// QuantumFeatureMap_71476845740520
// MI455X (gfx1250) — matrix operand order repaired and checked by reading, not compiled, not run on hardware
//
#include <hip/hip_runtime.h>
#include <hip/hip_bf16.h>
#include <math.h>

// ---------------------------------------------------------------------------
// QuantumFeatureMap on gfx1250:
//   out[b,w] = Q[b,:] (1x256) . W[:,w] (256x8)
//   Q[b, 16k+k'] = r_k r_k',  r_k = prod_i (bit_i(k) ? sin(a_i) : cos(a_i)),
//     a_i = (x[b,i]+1)*pi/4
//   W precomputed from params by simulating the fixed variational unitary on
//   the 16 reachable basis states (one tiny block), folding phases + Z signs.
// GEMM runs as split-f16 WMMA (QhiWhi + QhiWlo + QloWhi) with f32 accum,
// with Q and W staged in LDS directly in WMMA fragment layout so each
// v16h operand is one contiguous 32-byte LDS read (no VALU shuffling).
// ---------------------------------------------------------------------------

typedef __attribute__((ext_vector_type(16))) _Float16 v16h;
typedef __attribute__((ext_vector_type(8)))  float    v8f;

struct c2 { float x, y; };
__device__ __forceinline__ c2 cmul(c2 a, c2 b) { return {a.x*b.x - a.y*b.y, a.x*b.y + a.y*b.x}; }
__device__ __forceinline__ c2 cadd(c2 a, c2 b) { return {a.x + b.x, a.y + b.y}; }
__device__ __forceinline__ void mm2(const c2 A[2][2], const c2 B[2][2], c2 C[2][2]) {
#pragma unroll
  for (int i = 0; i < 2; ++i)
#pragma unroll
    for (int j = 0; j < 2; ++j)
      C[i][j] = cadd(cmul(A[i][0], B[0][j]), cmul(A[i][1], B[1][j]));
}

// ---------------------------------------------------------------------------
// Phase 1: build W[256][16] (cols 8..15 zero) from params. One block, 256 thr.
// S[j][c] = U[j, basis(c)] for the 16 reachable columns basis(c) = c<<4.
// ---------------------------------------------------------------------------
__global__ __launch_bounds__(256) void qfm_build_W(const float* __restrict__ params,
                                                   float* __restrict__ Wout) {
  __shared__ c2 S[256][16];  // 32 KB
  const int t = threadIdx.x;

  for (int idx = t; idx < 256 * 16; idx += 256) {
    int j = idx >> 4, c = idx & 15;
    S[j][c] = { (j == (c << 4)) ? 1.0f : 0.0f, 0.0f };
  }
  __syncthreads();

  for (int layer = 0; layer < 2; ++layer) {
    // single-qubit gates: G = RZ * RY * RX (RX applied first)
    for (int w = 0; w < 8; ++w) {
      float tx = 0.5f * params[(layer * 8 + w) * 3 + 0];
      float ty = 0.5f * params[(layer * 8 + w) * 3 + 1];
      float tz = 0.5f * params[(layer * 8 + w) * 3 + 2];
      float cx = cosf(tx), sx = sinf(tx);
      float cy = cosf(ty), sy = sinf(ty);
      float cz = cosf(tz), sz = sinf(tz);
      c2 RX[2][2] = { { {cx, 0.f}, {0.f, -sx} }, { {0.f, -sx}, {cx, 0.f} } };
      c2 RY[2][2] = { { {cy, 0.f}, {-sy, 0.f} }, { {sy, 0.f}, {cy, 0.f} } };
      c2 RZ[2][2] = { { {cz, -sz}, {0.f, 0.f} }, { {0.f, 0.f}, {cz, sz} } };
      c2 T[2][2], G[2][2];
      mm2(RY, RX, T);
      mm2(RZ, T, G);

      const int half = 1 << (7 - w);  // pair distance; wire 0 = MSB
      for (int task = t; task < 2048; task += 256) {
        int col = task & 15;
        int pr  = task >> 4;              // 0..127
        int xx  = pr >> (7 - w);
        int zz  = pr & (half - 1);
        int j0  = (xx << (8 - w)) + zz;
        int j1  = j0 + half;
        c2 a0 = S[j0][col], a1 = S[j1][col];
        S[j0][col] = cadd(cmul(G[0][0], a0), cmul(G[0][1], a1));
        S[j1][col] = cadd(cmul(G[1][0], a0), cmul(G[1][1], a1));
      }
      __syncthreads();
    }
    // CNOT chain: control c (bit 7-c), target c+1 (bit 6-c)
    for (int c = 0; c < 7; ++c) {
      const int td = 1 << (6 - c);
      for (int task = t; task < 1024; task += 256) {
        int col = task & 15;
        int pr  = task >> 4;              // 0..63
        int xx  = pr >> (6 - c);
        int zz  = pr & (td - 1);
        int j0  = (xx << (8 - c)) + (1 << (7 - c)) + zz;  // control=1, target=0
        int j1  = j0 + td;
        c2 a0 = S[j0][col], a1 = S[j1][col];
        S[j0][col] = a1;
        S[j1][col] = a0;
      }
      __syncthreads();
    }
  }

  // A[p][w] = Re( i^((popc(k')-popc(k))&3) * sum_j sign_w(j) U[j,k] conj(U[j,k']) )
  const int k  = t >> 4;
  const int k2 = t & 15;
  const int d  = (__popc((unsigned)k2) - __popc((unsigned)k)) & 3;
  float acc[8];
#pragma unroll
  for (int w = 0; w < 8; ++w) acc[w] = 0.0f;

  for (int j = 0; j < 256; ++j) {
    c2 u = S[j][k];
    c2 v = S[j][k2];
    float re = u.x * v.x + u.y * v.y;   // Re(u * conj(v))
    float im = u.y * v.x - u.x * v.y;   // Im(u * conj(v))
    float comp = (d == 0) ? re : (d == 1) ? -im : (d == 2) ? -re : im;
#pragma unroll
    for (int w = 0; w < 8; ++w) {
      float sgn = ((j >> (7 - w)) & 1) ? -1.0f : 1.0f;
      acc[w] += sgn * comp;
    }
  }
#pragma unroll
  for (int w = 0; w < 8; ++w)  Wout[t * 16 + w] = acc[w];
#pragma unroll
  for (int w = 8; w < 16; ++w) Wout[t * 16 + w] = 0.0f;
}

// ---------------------------------------------------------------------------
// Phase 2: out[65536][8] = Q[65536][256] * W[256][16] via split-f16 WMMA.
// Block = 64 threads = 2 waves; each wave owns a 16-sample M-tile.
//
// Fragment-layout staging:
//   A (16-bit 16x32, ISA 7.12.2): lane group ga=lane>>4 reads, per chunk kc,
//     K = kc*32 + ga*8 + (e<8 ? e : e+8)   for element e of v16h
//   B (32x16): lane group gb reads K = kc*32 + gb*8 + (e<8 ? e : e+8)
// Inverse maps (writer side):
//   A: kc=K>>5, ga=(K>>3)&1, e=((K>>4)&1)*8 + (K&7)
//   B: kc=K>>5, gb=(K>>3)&1, e=((K>>4)&1)*8 + (K&7)
// ---------------------------------------------------------------------------
__global__ __launch_bounds__(64) void qfm_gemm(const float* __restrict__ x,
                                               const float* __restrict__ Wmat,
                                               float* __restrict__ out) {
  // [wave][split(hi/lo)][group][m][kc] -> one v16h fragment each: 32 KB
  __shared__ v16h Afrag[2][2][2][16][8];
  // [split(hi/lo)][group][n][kc] -> one v16h fragment each: 16 KB
  __shared__ v16h Bfrag[2][2][16][8];

  const int t    = threadIdx.x;
  const int wave = t >> 5;
  const int lane = t & 31;

  // ---- Stage W into B-fragment layout (split f16), once per block ----
  for (int idx = t; idx < 256 * 16; idx += 64) {
    int K = idx >> 4, n = idx & 15;
    float wv = Wmat[K * 16 + n];
    _Float16 hi = (_Float16)wv;
    _Float16 lo = (_Float16)(wv - (float)hi);
    int kc = K >> 5, gb = (K >> 3) & 1, e = ((K >> 4) & 1) * 8 + (K & 7);
    ((_Float16*)&Bfrag[0][gb][n][kc])[e] = hi;
    ((_Float16*)&Bfrag[1][gb][n][kc])[e] = lo;
  }

  // ---- Build this wave's Q tile directly in A-fragment layout ----
  const int tileBase = blockIdx.x * 32 + wave * 16;
  const int si = lane >> 1;   // sample row within tile
  const int hh = lane & 1;    // which 128-column half this lane fills
  const int sample = tileBase + si;

  float cs[4], sn[4];
#pragma unroll
  for (int i = 0; i < 4; ++i) {
    float ang = (x[sample * 4 + i] + 1.0f) * 0.7853981633974483f;  // (x+1)*pi/4
    cs[i] = cosf(ang);
    sn[i] = sinf(ang);
  }
  float r[16];
#pragma unroll
  for (int k = 0; k < 16; ++k) {
    float v = ((k >> 3) & 1) ? sn[0] : cs[0];
    v *= ((k >> 2) & 1) ? sn[1] : cs[1];
    v *= ((k >> 1) & 1) ? sn[2] : cs[2];
    v *= (k & 1)        ? sn[3] : cs[3];
    r[k] = v;
  }
  // Write K pairs (even/odd) so hi/lo half-stores merge into b32 stores.
  for (int ii = 0; ii < 64; ++ii) {
    int K0 = hh * 128 + ii * 2;           // even K
    float q0 = r[K0 >> 4] * r[K0 & 15];
    float q1 = r[K0 >> 4] * r[(K0 + 1) & 15];
    _Float16 hi0 = (_Float16)q0, hi1 = (_Float16)q1;
    _Float16 lo0 = (_Float16)(q0 - (float)hi0);
    _Float16 lo1 = (_Float16)(q1 - (float)hi1);
    int kc = K0 >> 5;
    int ga = (K0 >> 3) & 1;
    int e  = ((K0 >> 4) & 1) * 8 + (K0 & 7);   // even
    _Float16* ah = (_Float16*)&Afrag[wave][0][ga][si][kc];
    _Float16* al = (_Float16*)&Afrag[wave][1][ga][si][kc];
    ah[e] = hi0; ah[e + 1] = hi1;
    al[e] = lo0; al[e + 1] = lo1;
  }
  __syncthreads();

  // ---- WMMA GEMM: C(16x16) += A(16x32) * B(32x16) over 8 K-chunks ----
  const int g = lane >> 4;     // fragment group (same role for A and B here)
  const int m = lane & 15;     // A row / B column owned by this lane

  v8f accHH = {0.f, 0.f, 0.f, 0.f, 0.f, 0.f, 0.f, 0.f};
  v8f accHL = {0.f, 0.f, 0.f, 0.f, 0.f, 0.f, 0.f, 0.f};
  v8f accLH = {0.f, 0.f, 0.f, 0.f, 0.f, 0.f, 0.f, 0.f};

  for (int kc = 0; kc < 8; ++kc) {
    v16h ahi = Afrag[wave][0][g][m][kc];
    v16h alo = Afrag[wave][1][g][m][kc];
    v16h bhi = Bfrag[0][g][m][kc];
    v16h blo = Bfrag[1][g][m][kc];
    // Three independent accumulator chains -> the 3 WMMAs per chunk have no
    // mutual RAW hazard, hiding WMMA->WMMA latency.
    accHH = __builtin_amdgcn_wmma_f32_16x16x32_f16(false, ahi, false, bhi,
                                                   (short)0, accHH, false, false);
    accHL = __builtin_amdgcn_wmma_f32_16x16x32_f16(false, ahi, false, blo,
                                                   (short)0, accHL, false, false);
    accLH = __builtin_amdgcn_wmma_f32_16x16x32_f16(false, alo, false, bhi,
                                                   (short)0, accLH, false, false);
  }
  v8f accv = accHH + accHL + accLH;

  // C/D layout: VGPR rr -> M = rr (lanes 0-15) or rr+8 (lanes 16-31); N = lane&15.
  const int n = lane & 15;
  if (n < 8) {
    const int m0 = (lane < 16) ? 0 : 8;
#pragma unroll
    for (int rr = 0; rr < 8; ++rr) {
      out[(tileBase + m0 + rr) * 8 + n] = accv[rr];
    }
  }
}

// ---------------------------------------------------------------------------
extern "C" void kernel_launch(void* const* d_in, const int* in_sizes, int n_in,
                              void* d_out, int out_size, void* d_ws, size_t ws_size,
                              hipStream_t stream) {
  (void)in_sizes; (void)n_in; (void)out_size; (void)ws_size;
  const float* x      = (const float*)d_in[0];   // (65536, 4)
  const float* params = (const float*)d_in[1];   // (2, 8, 3)
  float* out = (float*)d_out;                    // (65536, 8)
  float* Wm  = (float*)d_ws;                     // 256*16 floats = 16 KB

  qfm_build_W<<<1, 256, 0, stream>>>(params, Wm);

  const int BATCH = 65536;
  const int blocks = BATCH / 32;                 // 2 waves * 16 samples per block
  qfm_gemm<<<blocks, 64, 0, stream>>>(x, Wm, out);
}
